// LearnedSimulator_61692910240021
// MI455X (gfx1250) — matrix operand order repaired and checked by reading, not compiled, not run on hardware
//
#include <hip/hip_runtime.h>
#include <math.h>

// ---------------------------------------------------------------------------
// CDNA5 / gfx1250 fused GNN simulator.
// All GEMMs run on v_wmma_f32_16x16x32_bf16 (wave32, f32 accumulate).
// Tile staging uses GLOBAL_LOAD_ASYNC_TO_LDS_B128 (ASYNCcnt) where the copy
// is conversion-free.  One fused kernel template: stage rows -> L1 -> ReLU ->
// L2 -> ReLU -> L3 -> LayerNorm -> mode-specific output (incl. fused
// segment_sum atomics and residual adds).
// ---------------------------------------------------------------------------

typedef __bf16 bf16;
typedef __attribute__((ext_vector_type(16))) __bf16 v16bf;
typedef __attribute__((ext_vector_type(8)))  __bf16 v8bf;
typedef __attribute__((ext_vector_type(8)))  float  v8f;

static constexpr int   NN     = 16384;
static constexpr int   EE     = 262144;
static constexpr float RADIUS = 0.015f;
static constexpr float DT     = 0.005f;
static constexpr float VSTD   = 0.0025f;

// input staging modes
static constexpr int IM_NODE_FEAT   = 0;  // features from position_sequence (K=16 pad 32)
static constexpr int IM_EDGE_FEAT   = 1;  // rel/dist features (K=3 pad 32)
static constexpr int IM_CONCAT_EDGE = 2;  // [e | x[s] | x[r]]  (K=384)
static constexpr int IM_CONCAT_NODE = 3;  // [x | agg]          (K=256)
static constexpr int IM_PLAIN       = 4;  // x rows             (K=128)
// output modes
static constexpr int OM_X_INIT = 0;  // LN -> xf(f32) + xb(bf16)
static constexpr int OM_E_INIT = 1;  // LN -> eb(bf16)
static constexpr int OM_E_UPD  = 2;  // LN -> atomic agg += ; eb += (residual)
static constexpr int OM_X_UPD  = 3;  // LN -> xf += ; xb = bf16(xf)
static constexpr int OM_H      = 4;  // (NL==2) bias+relu -> hb

#define DEVFN static __device__ __forceinline__

// 16-byte global -> LDS async copy (no VGPR data path, ASYNCcnt-tracked).
// On AMDGPU the low 32 bits of a generic pointer into LDS are the LDS byte
// address, which is what VDST of global_load_async_to_lds_* expects.
DEVFN void async_g2l_16(void* lds_ptr, const void* gptr) {
  unsigned l = (unsigned)(uintptr_t)lds_ptr;
  asm volatile("global_load_async_to_lds_b128 %0, %1, off"
               :: "v"(l), "v"(gptr) : "memory");
}
DEVFN void async_wait0() {
  asm volatile("s_wait_asynccnt 0" ::: "memory");
}

DEVFN void zero_acc(v8f c[8]) {
#pragma unroll
  for (int i = 0; i < 8; ++i)
#pragma unroll
    for (int j = 0; j < 8; ++j) c[i][j] = 0.f;
}

// One 16-row strip x 128 cols, K = KP (multiple of 32).
// A from LDS (row-major, padded stride sa), B from packed global fragments.
// A lane layout (16-bit A 16x32): lanes 0-15 row m hold K {0..15},
// lanes 16-31 row m hold K {16..31}.  B packed so each lane loads one
// contiguous v16bf (32B).
template <int KP>
DEVFN void gemm_tile(const bf16* abuf, int sa, int m0, int lane,
                     const v16bf* __restrict__ bp, v8f c[8]) {
  const int  half = lane >> 4;
  const bf16* arow = abuf + (m0 + (lane & 15)) * sa + half * 16;
  for (int kc = 0; kc < KP / 32; ++kc) {
    union { v16bf v; v8bf h[2]; } a;
    a.h[0] = *(const v8bf*)(arow + kc * 32);
    a.h[1] = *(const v8bf*)(arow + kc * 32 + 8);
#pragma unroll
    for (int nt = 0; nt < 8; ++nt) {
      v16bf b = bp[(nt * (KP / 32) + kc) * 32 + lane];
      c[nt] = __builtin_amdgcn_wmma_f32_16x16x32_bf16(
          false, a.v, false, b, (short)0, c[nt], false, false);
    }
  }
}

// bias + relu + bf16, scatter C layout (row = v + 8*(lane>>4), col = nt*16+lane&15)
// into row-major LDS activation buffer.
DEVFN void act_store(bf16* act, int sact, int m0, int lane,
                     const float* __restrict__ bias, v8f c[8]) {
  const int half = lane >> 4, n = lane & 15;
#pragma unroll
  for (int nt = 0; nt < 8; ++nt) {
    float bc = bias[nt * 16 + n];
#pragma unroll
    for (int v = 0; v < 8; ++v) {
      float x = fmaxf(c[nt][v] + bc, 0.f);
      act[(m0 + v + 8 * half) * sact + nt * 16 + n] = (bf16)x;
    }
  }
}

template <int KIN, int NL, int IMODE, int OMODE>
__global__ __launch_bounds__(256) void mlp3(
    const float* __restrict__ posq, const int* __restrict__ snd,
    const int* __restrict__ rcv,
    float* __restrict__ xf, bf16* __restrict__ xb,
    bf16* __restrict__ eb, float* __restrict__ agg, bf16* __restrict__ hb,
    const v16bf* __restrict__ B1, const float* __restrict__ b1,
    const v16bf* __restrict__ B2, const float* __restrict__ b2,
    const v16bf* __restrict__ B3, const float* __restrict__ b3,
    const float* __restrict__ lg, const float* __restrict__ lb) {
  constexpr int SA   = KIN + 8;   // padded bf16 stride (stride mod 64 DW != 0)
  constexpr int SACT = 136;
  extern __shared__ char smem[];
  int*  sIdx = (int*)smem;          // 128 senders
  int*  rIdx = sIdx + 128;          // 128 receivers
  bf16* xs   = (bf16*)(smem + 1024);
  bf16* act  = xs + 128 * SA;

  const int  tid  = threadIdx.x;
  const int  lane = tid & 31;
  const int  m0   = (tid >> 5) * 16;           // wave's row strip
  const long base = (long)blockIdx.x * 128;    // tile base row

  // ------------------- stage input tile into LDS (bf16) -------------------
  if constexpr (IMODE == IM_NODE_FEAT) {
    if (tid < 128) {
      const float* p  = posq + (base + tid) * 12;
      bf16*        xr = xs + tid * SA;
      float v[12];
#pragma unroll
      for (int i = 0; i < 12; ++i) { v[i] = p[i]; xr[i] = (bf16)v[i]; }
      const float inv = 1.f / RADIUS;
      float px = v[10], py = v[11];
      xr[12] = (bf16)fminf(1.f, fmaxf(-1.f, px * inv));
      xr[13] = (bf16)fminf(1.f, fmaxf(-1.f, py * inv));
      xr[14] = (bf16)fminf(1.f, fmaxf(-1.f, (1.f - px) * inv));
      xr[15] = (bf16)fminf(1.f, fmaxf(-1.f, (1.f - py) * inv));
#pragma unroll
      for (int i = 16; i < 32; ++i) xr[i] = (bf16)0.f;
    }
  } else if constexpr (IMODE == IM_EDGE_FEAT) {
    if (tid < 128) {
      long j = base + tid;
      int  s = snd[j], r = rcv[j];
      float sx = posq[(long)s * 12 + 10], sy = posq[(long)s * 12 + 11];
      float rx = posq[(long)r * 12 + 10], ry = posq[(long)r * 12 + 11];
      const float inv = 1.f / RADIUS;
      float ex = (sx - rx) * inv, ey = (sy - ry) * inv;
      float d  = sqrtf(ex * ex + ey * ey);
      bf16* xr = xs + tid * SA;
      xr[0] = (bf16)ex; xr[1] = (bf16)ey; xr[2] = (bf16)d;
#pragma unroll
      for (int i = 3; i < 32; ++i) xr[i] = (bf16)0.f;
    }
  } else if constexpr (IMODE == IM_CONCAT_EDGE) {
    if (tid < 128) { long j = base + tid; sIdx[tid] = snd[j]; rIdx[tid] = rcv[j]; }
    __syncthreads();
    for (int idx = tid; idx < 128 * 48; idx += 256) {   // 48 x 8-bf16 chunks/row
      int r = idx / 48, cch = idx % 48;
      int seg = cch >> 4, o = cch & 15;
      const v8bf* src;
      if (seg == 0)      src = (const v8bf*)(eb + (base + r) * 128) + o;
      else if (seg == 1) src = (const v8bf*)(xb + (long)sIdx[r] * 128) + o;
      else               src = (const v8bf*)(xb + (long)rIdx[r] * 128) + o;
      async_g2l_16(xs + r * SA + cch * 8, src);
    }
  } else if constexpr (IMODE == IM_CONCAT_NODE) {
    for (int idx = tid; idx < 128 * 32; idx += 256) {
      int r = idx / 32, cch = idx % 32;
      if (cch < 16) {
        async_g2l_16(xs + r * SA + cch * 8,
                     (const v8bf*)(xb + (base + r) * 128) + cch);
      } else {
        const float* s = agg + (base + r) * 128 + (cch - 16) * 8;
        float4 f0 = ((const float4*)s)[0];
        float4 f1 = ((const float4*)s)[1];
        v8bf o8;
        o8[0]=(bf16)f0.x; o8[1]=(bf16)f0.y; o8[2]=(bf16)f0.z; o8[3]=(bf16)f0.w;
        o8[4]=(bf16)f1.x; o8[5]=(bf16)f1.y; o8[6]=(bf16)f1.z; o8[7]=(bf16)f1.w;
        *(v8bf*)(xs + r * SA + cch * 8) = o8;
      }
    }
  } else {  // IM_PLAIN
    for (int idx = tid; idx < 128 * 16; idx += 256) {
      int r = idx / 16, cch = idx % 16;
      async_g2l_16(xs + r * SA + cch * 8,
                   (const v8bf*)(xb + (base + r) * 128) + cch);
    }
  }
  if constexpr (IMODE == IM_CONCAT_EDGE || IMODE == IM_CONCAT_NODE ||
                IMODE == IM_PLAIN) {
    async_wait0();   // all of this wave's async LDS writes have landed
  }
  __syncthreads();

  // ------------------------------ GEMM chain -------------------------------
  v8f c[8];
  zero_acc(c);
  gemm_tile<KIN>(xs, SA, m0, lane, B1, c);
  act_store(act, SACT, m0, lane, b1, c);
  __syncthreads();
  zero_acc(c);
  gemm_tile<128>(act, SACT, m0, lane, B2, c);

  const int half = lane >> 4, n = lane & 15;

  if constexpr (NL == 2) {  // decoder hidden: bias+relu -> hb
#pragma unroll
    for (int nt = 0; nt < 8; ++nt) {
      float bc = b2[nt * 16 + n];
#pragma unroll
      for (int v = 0; v < 8; ++v) {
        float x = fmaxf(c[nt][v] + bc, 0.f);
        hb[(base + m0 + v + 8 * half) * 128 + nt * 16 + n] = (bf16)x;
      }
    }
    return;
  }

  __syncthreads();                    // everyone done reading act
  act_store(act, SACT, m0, lane, b2, c);
  __syncthreads();
  zero_acc(c);
  gemm_tile<128>(act, SACT, m0, lane, B3, c);

  // bias3
#pragma unroll
  for (int nt = 0; nt < 8; ++nt) {
    float bc = b3[nt * 16 + n];
#pragma unroll
    for (int v = 0; v < 8; ++v) c[nt][v] += bc;
  }

  // LayerNorm over 128 cols: row lives across 16 lanes x 8 nt for each VGPR v.
  float mean[8], rstd[8];
#pragma unroll
  for (int v = 0; v < 8; ++v) {
    float s = 0.f, q = 0.f;
#pragma unroll
    for (int nt = 0; nt < 8; ++nt) { float x = c[nt][v]; s += x; q += x * x; }
#pragma unroll
    for (int m = 1; m < 16; m <<= 1) {
      s += __shfl_xor(s, m, 32);
      q += __shfl_xor(q, m, 32);
    }
    float mu = s * 0.0078125f;
    mean[v]  = mu;
    rstd[v]  = rsqrtf(q * 0.0078125f - mu * mu + 1e-5f);
  }

#pragma unroll
  for (int nt = 0; nt < 8; ++nt) {
    int   col = nt * 16 + n;
    float g = lg[col], bb = lb[col];
#pragma unroll
    for (int v = 0; v < 8; ++v) {
      int  rl  = m0 + v + 8 * half;
      long row = base + rl;
      float o  = (c[nt][v] - mean[v]) * rstd[v] * g + bb;
      if constexpr (OMODE == OM_X_INIT) {
        xf[row * 128 + col] = o;
        xb[row * 128 + col] = (bf16)o;
      } else if constexpr (OMODE == OM_E_INIT) {
        eb[row * 128 + col] = (bf16)o;
      } else if constexpr (OMODE == OM_E_UPD) {
        atomicAdd(agg + (long)rIdx[rl] * 128 + col, o);     // fused segment_sum
        float eo = (float)xs[rl * SA + col];                // old e from staging
        eb[row * 128 + col] = (bf16)(eo + o);               // fused residual
      } else if constexpr (OMODE == OM_X_UPD) {
        long gi = row * 128 + col;
        float xn = xf[gi] + o;
        xf[gi] = xn;
        xb[gi] = (bf16)xn;
      }
    }
  }
}

// Pack fp32 weight (kin x 128, row major) into WMMA B fragments, zero-padded
// to kp rows.  Fragment layout: [(nt*KC+kc)*32 + lane] x 16 bf16;
// lane<16 -> K kc*32+0..15, lane>=16 -> K kc*32+16..31; col = nt*16 + lane&15.
__global__ __launch_bounds__(256) void pack_w(const float* __restrict__ W,
                                              bf16* __restrict__ dst,
                                              int kin, int kp) {
  int idx = blockIdx.x * 256 + threadIdx.x;
  if (idx >= kp * 128) return;
  int e    = idx & 15;
  int lane = (idx >> 4) & 31;
  int fc   = idx >> 9;
  int KC   = kp >> 5;
  int kc   = fc % KC;
  int nt   = fc / KC;
  int half = lane >> 4, n = lane & 15;
  int k    = kc * 32 + half * 16 + e;
  int col  = nt * 16 + n;
  float v  = (k < kin) ? W[k * 128 + col] : 0.f;
  dst[idx] = (bf16)v;
}

__global__ __launch_bounds__(256) void zero_f32(float* __restrict__ p, int nel) {
  int i = blockIdx.x * 256 + threadIdx.x;
  if (i < nel) p[i] = 0.f;
}

// decoder layer 3 (128 -> 2) + output assembly
__global__ __launch_bounds__(256) void decode_out(
    const bf16* __restrict__ hb, const float* __restrict__ W3,
    const float* __restrict__ b3, const float* __restrict__ posq,
    float* __restrict__ out) {
  __shared__ float w[256];
  int tid = threadIdx.x;
  w[tid]  = W3[tid];
  __syncthreads();
  long i = (long)blockIdx.x * 256 + tid;
  const v8bf* h8 = (const v8bf*)(hb + i * 128);
  float o0 = b3[0], o1 = b3[1];
#pragma unroll
  for (int cch = 0; cch < 16; ++cch) {
    v8bf hv = h8[cch];
#pragma unroll
    for (int j = 0; j < 8; ++j) {
      float f = (float)hv[j];
      o0 += f * w[2 * (cch * 8 + j)];
      o1 += f * w[2 * (cch * 8 + j) + 1];
    }
  }
  float vx = o0 * VSTD, vy = o1 * VSTD;
  out[i * 2 + 0]                = posq[i * 12 + 10] + vx * DT;
  out[i * 2 + 1]                = posq[i * 12 + 11] + vy * DT;
  out[2 * (long)NN + i * 2 + 0] = vx;
  out[2 * (long)NN + i * 2 + 1] = vy;
}

// ---------------------------------------------------------------------------
// Param flat order (insertion order of setup_inputs):
//  0 position_sequence, 1 velocity_sequence (unused), 2 senders, 3 receivers,
//  4.. node_enc: W0,b0,W1,b1,W2,b2,ln_g,ln_b           (4..11)
//  12.. edge_enc: same                                  (12..19)
//  20 + 16*s .. : step s edge enc (8 arrays) then node enc (8 arrays)
//  100.. dec: W0,b0,W1,b1,W2,b2                         (100..105)
// ---------------------------------------------------------------------------
extern "C" void kernel_launch(void* const* d_in, const int* in_sizes, int n_in,
                              void* d_out, int out_size, void* d_ws,
                              size_t ws_size, hipStream_t stream) {
  const float* posq = (const float*)d_in[0];
  const int*   snd  = (const int*)d_in[2];
  const int*   rcv  = (const int*)d_in[3];
  auto F = [&](int i) { return (const float*)d_in[i]; };

  char*  ws  = (char*)d_ws;
  size_t off = 0;
  auto alloc = [&](size_t bytes) -> void* {
    void* r = ws + off;
    off += (bytes + 255) & ~(size_t)255;
    return r;
  };
  bf16*  xb  = (bf16*) alloc((size_t)NN * 128 * 2);
  float* xf  = (float*)alloc((size_t)NN * 128 * 4);
  bf16*  eb  = (bf16*) alloc((size_t)EE * 128 * 2);
  float* agg = (float*)alloc((size_t)NN * 128 * 4);
  bf16*  hb  = (bf16*) alloc((size_t)NN * 128 * 2);

  bf16* wp[40];
  int   np = 0;
  auto packW = [&](int idx, int kin, int kp) -> int {
    bf16* dst = (bf16*)alloc((size_t)kp * 128 * 2);
    int   tot = kp * 128;
    pack_w<<<(tot + 255) / 256, 256, 0, stream>>>(F(idx), dst, kin, kp);
    wp[np] = dst;
    return np++;
  };
  int ne1 = packW(4, 16, 32),  ne2 = packW(6, 128, 128),  ne3 = packW(8, 128, 128);
  int ee1 = packW(12, 3, 32),  ee2 = packW(14, 128, 128), ee3 = packW(16, 128, 128);
  int se1[5], se2[5], se3[5], sn1[5], sn2[5], sn3[5];
  for (int s = 0; s < 5; ++s) {
    int b  = 20 + 16 * s;
    se1[s] = packW(b,      384, 384);
    se2[s] = packW(b + 2,  128, 128);
    se3[s] = packW(b + 4,  128, 128);
    sn1[s] = packW(b + 8,  256, 256);
    sn2[s] = packW(b + 10, 128, 128);
    sn3[s] = packW(b + 12, 128, 128);
  }
  int d1 = packW(100, 128, 128), d2 = packW(102, 128, 128);

  auto lds = [](int kin) -> size_t {
    return 1024 + (size_t)128 * (kin + 8) * 2 + (size_t)128 * 136 * 2;
  };
#define BP(i) ((const v16bf*)wp[i])

  // encoders
  mlp3<32, 3, IM_NODE_FEAT, OM_X_INIT><<<NN / 128, 256, lds(32), stream>>>(
      posq, snd, rcv, xf, xb, eb, agg, hb,
      BP(ne1), F(5), BP(ne2), F(7), BP(ne3), F(9), F(10), F(11));
  mlp3<32, 3, IM_EDGE_FEAT, OM_E_INIT><<<EE / 128, 256, lds(32), stream>>>(
      posq, snd, rcv, xf, xb, eb, agg, hb,
      BP(ee1), F(13), BP(ee2), F(15), BP(ee3), F(17), F(18), F(19));

  // message-passing steps
  for (int s = 0; s < 5; ++s) {
    int b = 20 + 16 * s;
    zero_f32<<<(NN * 128) / 256, 256, 0, stream>>>(agg, NN * 128);
    mlp3<384, 3, IM_CONCAT_EDGE, OM_E_UPD><<<EE / 128, 256, lds(384), stream>>>(
        posq, snd, rcv, xf, xb, eb, agg, hb,
        BP(se1[s]), F(b + 1), BP(se2[s]), F(b + 3), BP(se3[s]), F(b + 5),
        F(b + 6), F(b + 7));
    mlp3<256, 3, IM_CONCAT_NODE, OM_X_UPD><<<NN / 128, 256, lds(256), stream>>>(
        posq, snd, rcv, xf, xb, eb, agg, hb,
        BP(sn1[s]), F(b + 9), BP(sn2[s]), F(b + 11), BP(sn3[s]), F(b + 13),
        F(b + 14), F(b + 15));
  }

  // decoder
  mlp3<128, 2, IM_PLAIN, OM_H><<<NN / 128, 256, lds(128), stream>>>(
      posq, snd, rcv, xf, xb, eb, agg, hb,
      BP(d1), F(101), BP(d2), F(103), nullptr, nullptr, nullptr, nullptr);
  decode_out<<<NN / 256, 256, 0, stream>>>(hb, F(104), F(105), posq,
                                           (float*)d_out);
  (void)in_sizes; (void)n_in; (void)out_size; (void)ws_size;
}
